// CrossAttention_16664473109002
// MI455X (gfx1250) — matrix operand order repaired and checked by reading, not compiled, not run on hardware
//
#include <hip/hip_runtime.h>

// ---------------------------------------------------------------------------
// CDNA5 (gfx1250) fused cross-attention pipeline.
// q/k/v produced by depthwise 3x3 conv, stored as bf16 (v transposed so the
// WMMA B-operand is K-contiguous). GEMMs use v_wmma_f32_16x16x32_bf16 with
// f32 accumulation. Softmax fused into GEMM1 epilogue (K=128 fits one tile
// row). Top-k(51 of 512) along channels via stable-rank counting. GEMM2
// fused with residual add. GEMM1: 2 M-tiles/wave (2 WMMA per B fetch);
// GEMM2: 4 N-tiles/wave (4 WMMA per A fetch). WGP-scope prefetch on the
// streamed operand.
// ---------------------------------------------------------------------------

typedef __bf16 v16bf __attribute__((ext_vector_type(16)));
typedef float  v8f   __attribute__((ext_vector_type(8)));
typedef unsigned short u16;
typedef u16 u16x8  __attribute__((ext_vector_type(8)));
typedef u16 u16x16 __attribute__((ext_vector_type(16)));

#define NB    16      // batch
#define CQ    512     // query channels
#define CK    128     // kv channels
#define HW    4096    // 64*64 spatial
#define NKEEP 51      // int(0.1*512)

__device__ __forceinline__ u16 f2bf(float f) {
    union { float f; unsigned u; } x; x.f = f;
    unsigned u = x.u + 0x7FFFu + ((x.u >> 16) & 1u);   // round-to-nearest-even
    return (u16)(u >> 16);
}

__device__ __forceinline__ v16bf mk_a(const u16* qa) {
    u16x8 alo = *(const u16x8*)qa;          // K = hi*16 + 0..7
    u16x8 ahi = *(const u16x8*)(qa + 8);    // K = hi*16 + 8..15
    u16x16 av;
#pragma unroll
    for (int i = 0; i < 8; ++i) { av[i] = alo[i]; av[i + 8] = ahi[i]; }
    return __builtin_bit_cast(v16bf, av);
}

// --------------------------- depthwise 3x3 conv ----------------------------
__global__ void __launch_bounds__(256)
dwconv_kernel(const float* __restrict__ x, const float* __restrict__ w,
              const float* __restrict__ bias, u16* __restrict__ out,
              int Cn, int transpose) {
    long idx = (long)blockIdx.x * 256 + threadIdx.x;
    long total = (long)NB * Cn * HW;
    if (idx >= total) return;
    int n = (int)(idx & 4095);
    int c = (int)((idx >> 12) % Cn);
    int b = (int)(idx / ((long)Cn << 12));
    int h = n >> 6, ww = n & 63;
    const float* xc = x + ((long)(b * Cn + c) << 12);
    const float* wc = w + c * 9;
    float s = bias[c];
#pragma unroll
    for (int dy = 0; dy < 3; ++dy) {
        int hh = h + dy - 1;
        if (hh < 0 || hh > 63) continue;
#pragma unroll
        for (int dx = 0; dx < 3; ++dx) {
            int cw = ww + dx - 1;
            if (cw < 0 || cw > 63) continue;
            s += xc[(hh << 6) + cw] * wc[dy * 3 + dx];
        }
    }
    long oi = transpose ? ((((long)b << 12) + n) * Cn + c) : idx;
    out[oi] = f2bf(s);
}

// ----------------------- GEMM1 (q @ k^T) + softmax -------------------------
// Block = 256 threads = 8 waves, computes a 32x128 tile of aw (2 M-tiles per
// wave sharing one B fragment), then row-softmax in LDS.
__global__ void __launch_bounds__(256)
gemm1_softmax_kernel(const u16* __restrict__ qbf, const u16* __restrict__ kbf,
                     float* __restrict__ aw_soft) {
    __shared__ float tile[32][128];
    int tid  = threadIdx.x;
    int lane = tid & 31, wave = tid >> 5;
    int hi = lane >> 4, lr = lane & 15;
    int m0 = blockIdx.x * 32;
    int b  = blockIdx.y;
    int n0 = wave * 16;

    const u16* qrow0 = qbf + ((long)(b * CQ + m0 + lr) << 12);   // A tile 0
    const u16* qrow1 = qrow0 + ((long)16 << 12);                 // A tile 1
    const u16* krow  = kbf + ((long)(b * CK + n0 + lr) << 12);   // B: col = lr

    v8f acc0 = {}, acc1 = {};
    for (int kc = 0; kc < 128; ++kc) {          // 4096 / 32
        int off = kc * 32;
        v16bf a0 = mk_a(qrow0 + off + hi * 16);
        v16bf a1 = mk_a(qrow1 + off + hi * 16);
        u16x16 bvv = *(const u16x16*)(krow + off + hi * 16);     // B: K=16*hi..+15
        __builtin_prefetch(krow + off + 512, 0, 3);              // WGP-scope, ~16 iters ahead
        v16bf bb = __builtin_bit_cast(v16bf, bvv);
        acc0 = __builtin_amdgcn_wmma_f32_16x16x32_bf16(
            false, a0, false, bb, (short)0, acc0, false, false);
        acc1 = __builtin_amdgcn_wmma_f32_16x16x32_bf16(
            false, a1, false, bb, (short)0, acc1, false, false);
    }
    // C layout: VGPR v -> row v + 8*hi, col = n0 + lr
#pragma unroll
    for (int v = 0; v < 8; ++v) {
        tile[v + 8 * hi][n0 + lr]      = acc0[v];
        tile[16 + v + 8 * hi][n0 + lr] = acc1[v];
    }
    __syncthreads();

    if (tid < 32) {                              // one thread per row
        const float scale = 0.08838834764831845f;   // 1/sqrt(128)
        float mx = -3.4e38f;
        for (int j = 0; j < 128; ++j) mx = fmaxf(mx, tile[tid][j]);
        float sum = 0.f;
        for (int j = 0; j < 128; ++j) {
            float e = __expf((tile[tid][j] - mx) * scale);
            tile[tid][j] = e; sum += e;
        }
        float inv = 1.f / sum;
        float* dst = aw_soft + ((long)(b * CQ + m0 + tid) << 7);
        for (int j = 0; j < 128; ++j) dst[j] = tile[tid][j] * inv;
    }
}

// ------------------- top-k (51 of 512) mask along channels -----------------
__global__ void __launch_bounds__(512)
topk_mask_kernel(const float* __restrict__ aw_soft, float* __restrict__ aw_out,
                 u16* __restrict__ aw_bf) {
    __shared__ float vals[512];
    int c = threadIdx.x;
    int k = blockIdx.x;
    int b = blockIdx.y;
    long oi = ((long)(b * CQ + c) << 7) + k;
    float v = aw_soft[oi];
    vals[c] = v;
    __syncthreads();
    int rank = 0;
    for (int j = 0; j < 512; ++j) {
        float o = vals[j];
        rank += (o > v) || (o == v && j < c);    // stable rank == lax.top_k order
    }
    float m = (rank < NKEEP) ? v : 0.f;
    aw_out[oi] = m;
    aw_bf[oi]  = f2bf(m);
}

// --------------------- GEMM2 (aw @ v) + residual add -----------------------
// Block = 256 threads = 8 waves; covers 16 channels x 512 spatial columns.
// Each wave: 4 N-tiles, 4 K-chunks -> 16 WMMA, one A fragment per chunk.
__global__ void __launch_bounds__(256)
gemm2_residual_kernel(const u16* __restrict__ awbf, const u16* __restrict__ vtbf,
                      const float* __restrict__ xq, float* __restrict__ out) {
    int tid  = threadIdx.x;
    int lane = tid & 31, wave = tid >> 5;
    int hi = lane >> 4, lr = lane & 15;
    int m0 = blockIdx.x * 16;
    int b  = blockIdx.y;
    int nb = blockIdx.z * 512 + wave * 64;

    const u16* arow  = awbf + ((long)(b * CQ + m0 + lr) << 7);
    const u16* vbase = vtbf + ((((long)b << 12) + nb + lr) << 7);  // col = nb+lr

    v8f c[4] = {{}, {}, {}, {}};
#pragma unroll
    for (int kc = 0; kc < 4; ++kc) {
        v16bf a = mk_a(arow + kc * 32 + hi * 16);
#pragma unroll
        for (int t = 0; t < 4; ++t) {
            u16x16 bt = *(const u16x16*)(vbase + ((long)t << 11) + kc * 32 + hi * 16);
            c[t] = __builtin_amdgcn_wmma_f32_16x16x32_bf16(
                false, a, false, __builtin_bit_cast(v16bf, bt),
                (short)0, c[t], false, false);
        }
    }
#pragma unroll
    for (int v = 0; v < 8; ++v) {
        int row = v + 8 * hi;
        long base = (((long)(b * CQ + m0 + row)) << 12) + nb + lr;
#pragma unroll
        for (int t = 0; t < 4; ++t) {
            long i = base + t * 16;
            out[i] = xq[i] + c[t][v];
        }
    }
}

// ---------------------------------------------------------------------------
extern "C" void kernel_launch(void* const* d_in, const int* in_sizes, int n_in,
                              void* d_out, int out_size, void* d_ws, size_t ws_size,
                              hipStream_t stream) {
    (void)in_sizes; (void)n_in; (void)out_size; (void)ws_size;
    const float* x_q  = (const float*)d_in[0];
    const float* x_kv = (const float*)d_in[1];
    const float* wq   = (const float*)d_in[2];
    const float* bq   = (const float*)d_in[3];
    const float* wk   = (const float*)d_in[4];
    const float* bk   = (const float*)d_in[5];
    const float* wvp  = (const float*)d_in[6];
    const float* bvp  = (const float*)d_in[7];

    // workspace layout (bf16 q/k/v + f32 softmax aw + bf16 masked aw)
    u16*   q_bf    = (u16*)d_ws;                               // 16*512*4096
    u16*   k_bf    = q_bf + (size_t)NB * CQ * HW;              // 16*128*4096
    u16*   v_t     = k_bf + (size_t)NB * CK * HW;              // transposed [B][N][K]
    float* aw_soft = (float*)(v_t + (size_t)NB * CK * HW);     // 16*512*128 f32
    u16*   aw_bf   = (u16*)(aw_soft + (size_t)NB * CQ * CK);   // 16*512*128 bf16

    float* out1   = (float*)d_out;                             // x_q + attn_out
    float* out_aw = out1 + (size_t)NB * CQ * HW;               // masked aw (f32)

    long nq = (long)NB * CQ * HW;
    long nk = (long)NB * CK * HW;
    dwconv_kernel<<<(nq + 255) / 256, 256, 0, stream>>>(x_q,  wq,  bq,  q_bf, CQ, 0);
    dwconv_kernel<<<(nk + 255) / 256, 256, 0, stream>>>(x_kv, wk,  bk,  k_bf, CK, 0);
    dwconv_kernel<<<(nk + 255) / 256, 256, 0, stream>>>(x_kv, wvp, bvp, v_t,  CK, 1);

    gemm1_softmax_kernel<<<dim3(CQ / 32, NB), 256, 0, stream>>>(q_bf, k_bf, aw_soft);
    topk_mask_kernel<<<dim3(CK, NB), 512, 0, stream>>>(aw_soft, out_aw, aw_bf);
    gemm2_residual_kernel<<<dim3(CQ / 16, NB, HW / 512), 256, 0, stream>>>(
        aw_bf, v_t, x_q, out1);
}
